// ContinuousTimeSSMCell_42279658062146
// MI455X (gfx1250) — matrix operand order repaired and checked by reading, not compiled, not run on hardware
//
#include <hip/hip_runtime.h>
#include <hip/hip_bf16.h>

// ---------------- Problem constants (from reference setup) ----------------
constexpr int B  = 256;
constexpr int T  = 512;
constexpr int D  = 256;
constexpr int H  = 512;
constexpr int BT = B * T;           // 131072

// ---------------- Types ----------------
typedef __bf16 bf16_t;
typedef __attribute__((ext_vector_type(16))) __bf16 v16bf;
typedef __attribute__((ext_vector_type(8)))  __bf16 v8bf;
typedef __attribute__((ext_vector_type(8)))  float  v8f;

// float -> bf16, round-to-nearest-even, via bit ops (no reliance on cvt legality)
__device__ __forceinline__ bf16_t f2bf(float f) {
    unsigned u = __builtin_bit_cast(unsigned, f);
    unsigned r = (u + 0x7FFFu + ((u >> 16) & 1u)) >> 16;
    unsigned short h = (unsigned short)r;
    return __builtin_bit_cast(bf16_t, h);
}

// ---------------- WMMA fragment loaders (gfx1250 wave32 layouts) ----------
// A fragment: 16x32 bf16, row-major source, tile origin at `A`, leading dim lda.
// lane<16: row=lane, K 0..7 -> elems 0..7, K 16..23 -> elems 8..15
// lane>=16: row=lane-16, K 8..15 -> elems 0..7, K 24..31 -> elems 8..15
__device__ __forceinline__ v16bf load_a_frag(const bf16_t* A, int lda, int lane) {
    int row = lane & 15;
    int kb  = (lane >> 4) << 3;            // 0 or 8
    const bf16_t* p = A + row * lda + kb;
    v8bf lo = *(const v8bf*)(p);
    v8bf hi = *(const v8bf*)(p + 16);
    v16bf r;
#pragma unroll
    for (int i = 0; i < 8; ++i) { r[i] = lo[i]; r[i + 8] = hi[i]; }
    return r;
}

// B fragment for C = A * W^T: B_op[k][n] = W[n][k], W row-major, tile origin at `W`
// lane<16: col=lane, K 0..7 -> elems 0..7, K 16..23 -> elems 8..15 ; lane>=16: col=lane-16, K 8..15, K 24..31
__device__ __forceinline__ v16bf load_bT_frag(const bf16_t* W, int ldw, int lane) {
    int col = lane & 15;
    int kb  = (lane >> 4) << 3;            // 0 or 8
    const bf16_t* p = W + col * ldw + kb;
    v8bf lo = *(const v8bf*)(p);
    v8bf hi = *(const v8bf*)(p + 16);
    v16bf r;
#pragma unroll
    for (int i = 0; i < 8; ++i) { r[i] = lo[i]; r[i + 8] = hi[i]; }
    return r;
}

#define WMMA_BF16(a, b, c) \
    __builtin_amdgcn_wmma_f32_16x16x32_bf16(false, (a), false, (b), (short)0, (c), false, false)

// ---------------- Prep kernels ----------------
__global__ __launch_bounds__(256) void k_prep_x(const float* __restrict__ x,
                                                const float* __restrict__ mask,
                                                bf16_t* __restrict__ Xbf, int n) {
    int i = blockIdx.x * 256 + threadIdx.x;
    if (i < n) Xbf[i] = f2bf(x[i] * mask[i]);
}

__global__ __launch_bounds__(256) void k_prep_small(
        const float* __restrict__ W1, const float* __restrict__ Bw,
        const float* __restrict__ Cw, const float* __restrict__ W2,
        const float* __restrict__ b1, const float* __restrict__ logA,
        const float* __restrict__ h_init,
        bf16_t* __restrict__ W1h_bf, bf16_t* __restrict__ Bw_bf,
        bf16_t* __restrict__ Cw_bf, float* __restrict__ w1c0,
        float* __restrict__ w2f, float* __restrict__ b1f,
        float* __restrict__ negA, float* __restrict__ hf,
        bf16_t* __restrict__ hbf, float* __restrict__ dot) {
    int i = blockIdx.x * 256 + threadIdx.x;
    if (i < H * H) {                       // W1[:,1:] -> bf16 (H x H)
        int r = i >> 9, c = i & (H - 1);
        W1h_bf[i] = f2bf(W1[r * (1 + H) + 1 + c]);
    }
    if (i < H * D) Bw_bf[i] = f2bf(Bw[i]);
    if (i < D * H) Cw_bf[i] = f2bf(Cw[i]);
    if (i < H) {
        w1c0[i] = W1[i * (1 + H)];
        w2f[i]  = W2[i];
        b1f[i]  = b1[i];
        negA[i] = -__expf(logA[i]);        // A_diag (negative)
    }
    if (i < B * H) {
        float h0 = h_init[i];
        hf[i]  = h0;
        hbf[i] = f2bf(h0);
    }
    if (i < 2 * B) dot[i] = 0.f;
}

// ---------------- Phase 1: U = (x*mask) @ B_w^T, fp32 out (BT x H) -------
__global__ __launch_bounds__(256) void k_gemm_u(const bf16_t* __restrict__ X,
                                                const bf16_t* __restrict__ Bw,
                                                float* __restrict__ U) {
    const int lane = threadIdx.x & 31;
    const int w    = threadIdx.x >> 5;
    const int row0 = blockIdx.x * 128 + w * 16;
    const int col0 = blockIdx.y * 64;
    v8f acc[4] = {};
    for (int k0 = 0; k0 < D; k0 += 32) {
        v16bf a = load_a_frag(X + (size_t)row0 * D + k0, D, lane);
#pragma unroll
        for (int c = 0; c < 4; ++c) {
            v16bf b = load_bT_frag(Bw + (size_t)(col0 + c * 16) * D + k0, D, lane);
            acc[c] = WMMA_BF16(a, b, acc[c]);
        }
    }
    const int n_lane = lane & 15;
    const int m_off  = (lane >> 4) << 3;
#pragma unroll
    for (int c = 0; c < 4; ++c)
#pragma unroll
        for (int e = 0; e < 8; ++e)
            U[(size_t)(row0 + m_off + e) * H + col0 + c * 16 + n_lane] = acc[c][e];
}

// ---------------- Phase 2a: hidden GEMM + fused silu + dot(W2) ------------
// hidden[b,n] = silu(gap[b]*w1c0[n] + b1[n] + sum_k h[b,k]*W1h[n,k])
// dot[b] += sum_n hidden[b,n] * W2[n]   (atomic, ping-pong buffer)
__global__ __launch_bounds__(256) void k_step_a(const bf16_t* __restrict__ hbf,
                                                const bf16_t* __restrict__ W1h,
                                                const float* __restrict__ w1c0,
                                                const float* __restrict__ b1f,
                                                const float* __restrict__ w2f,
                                                const float* __restrict__ dts,
                                                float* __restrict__ dot, int t) {
    const int lane = threadIdx.x & 31;
    const int w    = threadIdx.x >> 5;
    const int row0 = blockIdx.x * 128 + w * 16;   // over B=256
    const int col0 = blockIdx.y * 64;             // over H=512
    v8f acc[4] = {};
    for (int k0 = 0; k0 < H; k0 += 32) {
        v16bf a = load_a_frag(hbf + (size_t)row0 * H + k0, H, lane);
#pragma unroll
        for (int c = 0; c < 4; ++c) {
            v16bf b = load_bT_frag(W1h + (size_t)(col0 + c * 16) * H + k0, H, lane);
            acc[c] = WMMA_BF16(a, b, acc[c]);
        }
    }
    const int n_lane = lane & 15;
    const int m_off  = (lane >> 4) << 3;
    float gap[8], s[8];
#pragma unroll
    for (int e = 0; e < 8; ++e) {
        gap[e] = dts[(row0 + m_off + e) * T + t];
        s[e] = 0.f;
    }
#pragma unroll
    for (int c = 0; c < 4; ++c) {
        int n = col0 + c * 16 + n_lane;
        float wc0 = w1c0[n], bb = b1f[n], w2n = w2f[n];
#pragma unroll
        for (int e = 0; e < 8; ++e) {
            float v  = acc[c][e] + gap[e] * wc0 + bb;
            float sg = 1.f / (1.f + __expf(-v));
            s[e] += v * sg * w2n;                     // silu * W2
        }
    }
    // reduce across the 16 lanes sharing the same row set (masks < 16)
#pragma unroll
    for (int off = 8; off >= 1; off >>= 1)
#pragma unroll
        for (int e = 0; e < 8; ++e)
            s[e] += __shfl_xor(s[e], off, 32);
    if (n_lane == 0) {
#pragma unroll
        for (int e = 0; e < 8; ++e)
            atomicAdd(&dot[row0 + m_off + e], s[e]);
    }
}

// ---------------- Phase 2b: h update (elementwise over B*H) ---------------
__global__ __launch_bounds__(256) void k_step_b(const float* __restrict__ dotbuf,
                                                float* __restrict__ dot_next,
                                                const float* __restrict__ b2,
                                                const float* __restrict__ negA,
                                                const float* __restrict__ U,
                                                float* __restrict__ hf,
                                                bf16_t* __restrict__ hbf,
                                                bf16_t* __restrict__ Hall, int t) {
    int i = blockIdx.x * 256 + threadIdx.x;        // B*H = 131072
    int b = i >> 9, j = i & (H - 1);
    float z     = dotbuf[b] + b2[0];
    float delta = (z > 20.f) ? z : log1pf(__expf(z));   // softplus
    float ex    = negA[j] * delta;
    ex          = fminf(0.f, fmaxf(-20.f, ex));
    float Abar  = __expf(ex);
    size_t m    = (size_t)(b * T + t);
    float hn    = Abar * hf[i] + U[m * H + j] * delta;
    hn          = fminf(50.f, fmaxf(-50.f, hn));
    hf[i]  = hn;
    hbf[i] = f2bf(hn);
    Hall[m * H + j] = f2bf(hn);
    if (i < B) dot_next[i] = 0.f;                  // clear other ping-pong buf
}

// ---------------- Phase 3: Y = Hall @ C_w^T + D*x (pre-LN) ----------------
__global__ __launch_bounds__(256) void k_gemm_y(const bf16_t* __restrict__ Hall,
                                                const bf16_t* __restrict__ Cw,
                                                const float* __restrict__ x,
                                                const float* __restrict__ mask,
                                                const float* __restrict__ Dp,
                                                float* __restrict__ out) {
    const int lane = threadIdx.x & 31;
    const int w    = threadIdx.x >> 5;
    const int row0 = blockIdx.x * 128 + w * 16;   // over BT
    const int col0 = blockIdx.y * 64;             // over D=256
    v8f acc[4] = {};
    for (int k0 = 0; k0 < H; k0 += 32) {
        v16bf a = load_a_frag(Hall + (size_t)row0 * H + k0, H, lane);
#pragma unroll
        for (int c = 0; c < 4; ++c) {
            v16bf b = load_bT_frag(Cw + (size_t)(col0 + c * 16) * H + k0, H, lane);
            acc[c] = WMMA_BF16(a, b, acc[c]);
        }
    }
    const int n_lane = lane & 15;
    const int m_off  = (lane >> 4) << 3;
#pragma unroll
    for (int c = 0; c < 4; ++c)
#pragma unroll
        for (int e = 0; e < 8; ++e) {
            size_t m = (size_t)(row0 + m_off + e);
            int dcol = col0 + c * 16 + n_lane;
            size_t idx = m * D + dcol;
            float xm = x[idx] * mask[idx];
            out[idx] = acc[c][e] + Dp[dcol] * xm;
        }
}

// ---------------- Phase 3b: LayerNorm over D=256, one wave per row --------
__global__ __launch_bounds__(256) void k_ln(float* __restrict__ y,
                                            const float* __restrict__ gamma,
                                            const float* __restrict__ beta) {
    int gid  = blockIdx.x * 256 + threadIdx.x;
    int row  = gid >> 5;
    int lane = gid & 31;
    if (row >= BT) return;
    float* r = y + (size_t)row * D;
    float v[8], s = 0.f, s2 = 0.f;
#pragma unroll
    for (int e = 0; e < 8; ++e) {
        v[e] = r[lane + e * 32];
        s += v[e]; s2 += v[e] * v[e];
    }
#pragma unroll
    for (int off = 16; off >= 1; off >>= 1) {
        s  += __shfl_xor(s,  off, 32);
        s2 += __shfl_xor(s2, off, 32);
    }
    float mu  = s * (1.f / D);
    float var = s2 * (1.f / D) - mu * mu;
    float rs  = rsqrtf(var + 1e-5f);
#pragma unroll
    for (int e = 0; e < 8; ++e) {
        int d = lane + e * 32;
        r[d] = (v[e] - mu) * rs * gamma[d] + beta[d];
    }
}

__global__ __launch_bounds__(256) void k_hfinal(const float* __restrict__ hf,
                                                float* __restrict__ out_tail) {
    int i = blockIdx.x * 256 + threadIdx.x;
    if (i < B * H) out_tail[i] = hf[i];
}

// ---------------- Host launcher ----------------
extern "C" void kernel_launch(void* const* d_in, const int* in_sizes, int n_in,
                              void* d_out, int out_size, void* d_ws, size_t ws_size,
                              hipStream_t stream) {
    (void)in_sizes; (void)n_in; (void)out_size; (void)ws_size;
    const float* x     = (const float*)d_in[0];
    const float* dts   = (const float*)d_in[1];
    const float* mask  = (const float*)d_in[2];
    const float* h_in  = (const float*)d_in[3];
    const float* logA  = (const float*)d_in[4];
    const float* Bw    = (const float*)d_in[5];
    const float* Cw    = (const float*)d_in[6];
    const float* Dp    = (const float*)d_in[7];
    const float* W1    = (const float*)d_in[8];
    const float* b1    = (const float*)d_in[9];
    const float* W2    = (const float*)d_in[10];
    const float* b2    = (const float*)d_in[11];
    const float* lng   = (const float*)d_in[12];
    const float* lnb   = (const float*)d_in[13];
    float* out = (float*)d_out;

    // workspace carve-out
    char* w = (char*)d_ws;
    auto alloc = [&](size_t bytes) -> void* {
        void* p = (void*)w;
        w += (bytes + 255) & ~(size_t)255;
        return p;
    };
    float*  U      = (float*)alloc((size_t)BT * H * 4);   // pre-delta B_bar
    bf16_t* Hall   = (bf16_t*)alloc((size_t)BT * H * 2);  // all h_t, bf16
    bf16_t* Xbf    = (bf16_t*)alloc((size_t)BT * D * 2);  // masked x, bf16
    bf16_t* W1h_bf = (bf16_t*)alloc((size_t)H * H * 2);
    bf16_t* Bw_bf  = (bf16_t*)alloc((size_t)H * D * 2);
    bf16_t* Cw_bf  = (bf16_t*)alloc((size_t)D * H * 2);
    float*  hf     = (float*)alloc((size_t)B * H * 4);
    bf16_t* hbf    = (bf16_t*)alloc((size_t)B * H * 2);
    float*  w1c0   = (float*)alloc(H * 4);
    float*  w2f    = (float*)alloc(H * 4);
    float*  b1f    = (float*)alloc(H * 4);
    float*  negA   = (float*)alloc(H * 4);
    float*  dot    = (float*)alloc(2 * B * 4);            // ping-pong

    // prep
    k_prep_x<<<(BT * D + 255) / 256, 256, 0, stream>>>(x, mask, Xbf, BT * D);
    {
        int mx = H * (1 + H);                              // 262656 covers all
        k_prep_small<<<(mx + 255) / 256, 256, 0, stream>>>(
            W1, Bw, Cw, W2, b1, logA, h_in,
            W1h_bf, Bw_bf, Cw_bf, w1c0, w2f, b1f, negA, hf, hbf, dot);
    }

    // Phase 1: U = Xbf @ Bw^T   (BT x H, K=D)
    k_gemm_u<<<dim3(BT / 128, H / 64), 256, 0, stream>>>(Xbf, Bw_bf, U);

    // Phase 2: serial scan over T
    for (int t = 0; t < T; ++t) {
        k_step_a<<<dim3(B / 128, H / 64), 256, 0, stream>>>(
            hbf, W1h_bf, w1c0, b1f, w2f, dts, dot + (t & 1) * B, t);
        k_step_b<<<(B * H) / 256, 256, 0, stream>>>(
            dot + (t & 1) * B, dot + ((t + 1) & 1) * B, b2, negA, U, hf, hbf, Hall, t);
    }

    // Phase 3: Y = Hall @ Cw^T + D*x  -> d_out, then LayerNorm in place
    k_gemm_y<<<dim3(BT / 128, D / 64), 256, 0, stream>>>(Hall, Cw_bf, x, mask, Dp, out);
    k_ln<<<(BT * 32 + 255) / 256, 256, 0, stream>>>(out, lng, lnb);
    k_hfinal<<<(B * H + 255) / 256, 256, 0, stream>>>(hf, out + (size_t)BT * D);
}
